// DeepSeekStyleLM_80685255623339
// MI455X (gfx1250) — matrix operand order repaired and checked by reading, not compiled, not run on hardware
//
#include <hip/hip_runtime.h>
#include <hip/hip_bf16.h>

// ---------------- problem constants ----------------
static constexpr int B_  = 1;
static constexpr int T_  = 2048;
static constexpr int C_  = 1024;
static constexpr int H_  = 16;
static constexpr int HKV_ = 8;
static constexpr int DH_ = 64;
static constexpr int L_  = 4;
static constexpr int V_  = 32000;
static constexpr int HFF_ = 4096;
static constexpr int KVD_ = HKV_ * DH_;   // 512

typedef __attribute__((ext_vector_type(16))) __bf16 v16bf;
typedef __attribute__((ext_vector_type(8)))  float  v8f;

__device__ inline __bf16 f2bf(float f) {
  unsigned u = __float_as_uint(f);
  u += 0x7fffu + ((u >> 16) & 1u);          // round to nearest even
  unsigned short h = (unsigned short)(u >> 16);
  return __builtin_bit_cast(__bf16, h);
}

__device__ inline v8f wmma_bf16(v16bf a, v16bf b, v8f c) {
  return __builtin_amdgcn_wmma_f32_16x16x32_bf16(
      /*neg_a=*/false, a, /*neg_b=*/false, b,
      /*c_mod=*/(short)0, c, /*reuse_a=*/false, /*reuse_b=*/false);
}

// ---------------- embedding gather ----------------
__global__ void __launch_bounds__(256) embed_kernel(
    const int* __restrict__ tokens, const float* __restrict__ emb,
    float* __restrict__ x) {
  int idx = blockIdx.x * 256 + threadIdx.x;           // T_*C_ threads
  int t = idx / C_, c = idx % C_;
  x[idx] = emb[(size_t)tokens[t] * C_ + c];
}

// ---------------- RMSNorm (one block per row) ----------------
__global__ void __launch_bounds__(256) rmsnorm_kernel(
    const float* __restrict__ x, const float* __restrict__ w,
    float* __restrict__ out) {
  __shared__ float red[256];
  int t = blockIdx.x;
  const float* row = x + (size_t)t * C_;
  float s = 0.f;
  for (int c = threadIdx.x; c < C_; c += 256) { float v = row[c]; s += v * v; }
  red[threadIdx.x] = s;
  __syncthreads();
  for (int off = 128; off > 0; off >>= 1) {
    if (threadIdx.x < off) red[threadIdx.x] += red[threadIdx.x + off];
    __syncthreads();
  }
  float inv = rsqrtf(red[0] / (float)C_ + 1e-5f);
  for (int c = threadIdx.x; c < C_; c += 256)
    out[(size_t)t * C_ + c] = row[c] * inv * w[c];
}

// ---------------- RoPE (in place on [T, nh, 64]) ----------------
__global__ void __launch_bounds__(256) rope_kernel(float* buf, int nh) {
  int idx = blockIdx.x * 256 + threadIdx.x;           // T_*nh*32 threads
  int per = nh * 32;
  int t = idx / per;
  int r = idx % per;
  int hh = r / 32, d = r % 32;
  size_t base = (size_t)t * nh * 64 + (size_t)hh * 64 + d;
  float x1 = buf[base], x2 = buf[base + 32];
  float inv = __powf(10000.f, -(float)d / 32.f);
  float ang = (float)t * inv;
  float sn = __sinf(ang), cs = __cosf(ang);
  buf[base]      = x1 * cs - x2 * sn;
  buf[base + 32] = x1 * sn + x2 * cs;
}

// ---------------- SwiGLU combine: fa = fb * silu(fa) ----------------
__global__ void __launch_bounds__(256) swiglu_kernel(
    float* __restrict__ fa, const float* __restrict__ fb) {
  int idx = blockIdx.x * 256 + threadIdx.x;           // T_*HFF_ threads
  float a = fa[idx], b = fb[idx];
  fa[idx] = b * a * (1.f / (1.f + __expf(-a)));
}

// ---------------- tiled bf16 WMMA GEMM: out = A[M,K] * W[N,K]^T (+res) ----
// Block tile 128x128, BK=32, 8 waves; each wave owns a 32x64 output region
// (2 A-frags x 4 B-frags -> 8 WMMAs per k-step). Exact tiling, no bounds
// checks (all shapes divide: M=2048, N in {512,1024,4096,32000}, K%32==0).
__global__ void __launch_bounds__(256) gemm_nt_kernel(
    const float* __restrict__ A, const float* __restrict__ W,
    const float* residual, float* out, int M, int N, int K) {
  __shared__ __bf16 As[128][36];
  __shared__ __bf16 Bs[128][36];

  const int m0 = blockIdx.y * 128;
  const int n0 = blockIdx.x * 128;
  const int tid = threadIdx.x;
  const int wv = tid >> 5;
  const int lane = tid & 31;
  const int l15 = lane & 15;
  const int khf = lane >> 4;
  const int rr = wv & 3;          // 32-row region (2 m-tiles)
  const int cc = wv >> 2;         // 64-col region (4 n-tiles)

  v8f acc[2][4];
#pragma unroll
  for (int mt = 0; mt < 2; ++mt)
#pragma unroll
    for (int j = 0; j < 4; ++j)
#pragma unroll
      for (int i = 0; i < 8; ++i) acc[mt][j][i] = 0.f;

  const int srow = tid >> 1;           // 0..127
  const int scol = (tid & 1) * 16;     // 0 or 16

  for (int k0 = 0; k0 < K; k0 += 32) {
    // cooperative stage: 128x32 fp32 -> bf16 for A and W (16 elems/thread)
    const float* pa = A + (size_t)(m0 + srow) * K + k0 + scol;
    const float* pw = W + (size_t)(n0 + srow) * K + k0 + scol;
#pragma unroll
    for (int q = 0; q < 4; ++q) {
      float4 av = ((const float4*)pa)[q];
      float4 wvv = ((const float4*)pw)[q];
      As[srow][scol + q * 4 + 0] = f2bf(av.x);
      As[srow][scol + q * 4 + 1] = f2bf(av.y);
      As[srow][scol + q * 4 + 2] = f2bf(av.z);
      As[srow][scol + q * 4 + 3] = f2bf(av.w);
      Bs[srow][scol + q * 4 + 0] = f2bf(wvv.x);
      Bs[srow][scol + q * 4 + 1] = f2bf(wvv.y);
      Bs[srow][scol + q * 4 + 2] = f2bf(wvv.z);
      Bs[srow][scol + q * 4 + 3] = f2bf(wvv.w);
    }
    // prefetch next K tile into cache while this tile computes
    if (k0 + 32 < K) {
      __builtin_prefetch(pa + 32, 0, 1);
      __builtin_prefetch(pw + 32, 0, 1);
    }
    __syncthreads();

    // A fragments (16x32, interleaved-by-8 across lane halves)
    v16bf af[2];
#pragma unroll
    for (int mt = 0; mt < 2; ++mt) {
      const int arow = rr * 32 + mt * 16 + l15;
#pragma unroll
      for (int e = 0; e < 8; ++e) {
        af[mt][e]     = As[arow][khf * 8 + e];
        af[mt][e + 8] = As[arow][16 + khf * 8 + e];
      }
    }
#pragma unroll
    for (int j = 0; j < 4; ++j) {
      // B fragment (32x16): column = weight row, K interleaved-by-8 across lane halves
      v16bf bf_;
      const int brow = cc * 64 + j * 16 + l15;
#pragma unroll
      for (int e = 0; e < 16; ++e) bf_[e] = Bs[brow][(e < 8) ? (khf * 8 + e) : (16 + khf * 8 + e - 8)];
      acc[0][j] = wmma_bf16(af[0], bf_, acc[0][j]);
      acc[1][j] = wmma_bf16(af[1], bf_, acc[1][j]);
    }
    __syncthreads();
  }

  // epilogue (C layout: element i -> row i + 8*half, col = lane&15),
  // residual branch hoisted out of the store loops
  if (residual) {
#pragma unroll
    for (int mt = 0; mt < 2; ++mt)
#pragma unroll
      for (int j = 0; j < 4; ++j)
#pragma unroll
        for (int i = 0; i < 8; ++i) {
          int gm = m0 + rr * 32 + mt * 16 + i + khf * 8;
          int gn = n0 + cc * 64 + j * 16 + l15;
          size_t o = (size_t)gm * N + gn;
          out[o] = acc[mt][j][i] + residual[o];
        }
  } else {
#pragma unroll
    for (int mt = 0; mt < 2; ++mt)
#pragma unroll
      for (int j = 0; j < 4; ++j)
#pragma unroll
        for (int i = 0; i < 8; ++i) {
          int gm = m0 + rr * 32 + mt * 16 + i + khf * 8;
          int gn = n0 + cc * 64 + j * 16 + l15;
          out[(size_t)gm * N + gn] = acc[mt][j][i];
        }
  }
}

// ---------------- flash attention (WMMA QK^T and PV) ----------------
// grid = (T/64, H), block = 128 (4 waves, one 16-row Q tile per wave)
__global__ void __launch_bounds__(128) attn_kernel(
    const float* __restrict__ qb, const float* __restrict__ kb,
    const float* __restrict__ vb, float* __restrict__ yb) {
  __shared__ __bf16 Pl[4][16][36];

  const int wv = threadIdx.x >> 5;
  const int lane = threadIdx.x & 31;
  const int l15 = lane & 15;
  const int khf = lane >> 4;
  const int qt = blockIdx.x * 4 + wv;
  const int q0 = qt * 16;
  const int h = blockIdx.y;
  const int kvh = h >> 1;                 // GQA group = 2

  // Q fragments (scale 1/sqrt(64) folded in)
  v16bf qa[2];
#pragma unroll
  for (int kc = 0; kc < 2; ++kc) {
    const float* qr = qb + (size_t)(q0 + l15) * C_ + h * 64 + kc * 32;
#pragma unroll
    for (int e = 0; e < 8; ++e) {
      qa[kc][e]     = f2bf(qr[khf * 8 + e] * 0.125f);
      qa[kc][e + 8] = f2bf(qr[16 + khf * 8 + e] * 0.125f);
    }
  }

  v8f o[4];
  float m_arr[8], l_arr[8];
#pragma unroll
  for (int j = 0; j < 4; ++j)
#pragma unroll
    for (int i = 0; i < 8; ++i) o[j][i] = 0.f;
#pragma unroll
  for (int i = 0; i < 8; ++i) { m_arr[i] = -3.0e38f; l_arr[i] = 0.f; }

  const int nkv = q0 + 16;                 // causal extent
  for (int kv0 = 0; kv0 < nkv; kv0 += 32) {
    // S = Q * K^T for 32 kv columns (2 n-subtiles x 2 k-chunks)
    v8f s[2];
#pragma unroll
    for (int c = 0; c < 2; ++c) {
#pragma unroll
      for (int i = 0; i < 8; ++i) s[c][i] = 0.f;
      const int ncol = kv0 + c * 16 + l15;
#pragma unroll
      for (int kc = 0; kc < 2; ++kc) {
        v16bf kf;
        const float* kr = kb + (size_t)ncol * KVD_ + kvh * 64 + kc * 32;
#pragma unroll
        for (int e = 0; e < 16; ++e) kf[e] = f2bf(kr[(e < 8) ? (khf * 8 + e) : (16 + khf * 8 + e - 8)]);
        s[c] = wmma_bf16(qa[kc], kf, s[c]);
      }
    }

    // online softmax + stage P into LDS (A-fragment via transpose)
#pragma unroll
    for (int i = 0; i < 8; ++i) {
      int row = q0 + i + khf * 8;
      float s0 = s[0][i]; if (kv0 + l15 > row) s0 = -3.0e38f;
      float s1 = s[1][i]; if (kv0 + 16 + l15 > row) s1 = -3.0e38f;
      float mloc = fmaxf(s0, s1);
#pragma unroll
      for (int off = 1; off < 16; off <<= 1)
        mloc = fmaxf(mloc, __shfl_xor(mloc, off, 32));
      float mnew = fmaxf(m_arr[i], mloc);
      float al = __expf(m_arr[i] - mnew);
      float p0 = __expf(s0 - mnew);
      float p1 = __expf(s1 - mnew);
      float rs = p0 + p1;
#pragma unroll
      for (int off = 1; off < 16; off <<= 1) rs += __shfl_xor(rs, off, 32);
      l_arr[i] = l_arr[i] * al + rs;
      m_arr[i] = mnew;
#pragma unroll
      for (int j = 0; j < 4; ++j) o[j][i] *= al;
      Pl[wv][i + khf * 8][l15]      = f2bf(p0);
      Pl[wv][i + khf * 8][16 + l15] = f2bf(p1);
    }

    // wave-private LDS transpose: drain DS before re-reading (no block
    // barrier allowed — kv trip count differs per wave)
    asm volatile("s_wait_dscnt 0" ::: "memory");

    v16bf pa;
#pragma unroll
    for (int e = 0; e < 8; ++e) {
      pa[e]     = Pl[wv][l15][khf * 8 + e];
      pa[e + 8] = Pl[wv][l15][16 + khf * 8 + e];
    }

    // O += P(16x32) * V(32x64)
#pragma unroll
    for (int j = 0; j < 4; ++j) {
      v16bf vf;
#pragma unroll
      for (int e = 0; e < 16; ++e) {
        int kk = (e < 8) ? (khf * 8 + e) : (16 + khf * 8 + e - 8);
        vf[e] = f2bf(vb[(size_t)(kv0 + kk) * KVD_ + kvh * 64 + j * 16 + l15]);
      }
      o[j] = wmma_bf16(pa, vf, o[j]);
    }
  }

  // normalize + write
#pragma unroll
  for (int j = 0; j < 4; ++j)
#pragma unroll
    for (int i = 0; i < 8; ++i) {
      int row = q0 + i + khf * 8;
      yb[(size_t)row * C_ + h * 64 + j * 16 + l15] = o[j][i] / l_arr[i];
    }
}

// ---------------- host orchestration ----------------
extern "C" void kernel_launch(void* const* d_in, const int* in_sizes, int n_in,
                              void* d_out, int out_size, void* d_ws, size_t ws_size,
                              hipStream_t stream) {
  (void)in_sizes; (void)n_in; (void)out_size; (void)ws_size;
  const int*   tokens      = (const int*)d_in[0];
  const float* emb         = (const float*)d_in[1];
  const float* attn_norm_w = (const float*)d_in[2];
  const float* Wq          = (const float*)d_in[3];
  const float* Wk          = (const float*)d_in[4];
  const float* Wv          = (const float*)d_in[5];
  const float* Wo          = (const float*)d_in[6];
  const float* ff_norm_w   = (const float*)d_in[7];
  const float* W1          = (const float*)d_in[8];
  const float* W2          = (const float*)d_in[9];
  const float* W3          = (const float*)d_in[10];
  const float* norm_f_w    = (const float*)d_in[11];
  float* out = (float*)d_out;

  float* ws = (float*)d_ws;
  const size_t M1 = 1024 * 1024;
  float* x  = ws;              // [T,C]   2M floats
  float* xn = ws + 2  * M1;    // [T,C]
  float* q  = ws + 4  * M1;    // [T,C]
  float* kb = ws + 6  * M1;    // [T,512]
  float* vb = ws + 7  * M1;    // [T,512]
  float* y  = ws + 8  * M1;    // [T,C]
  float* fa = ws + 10 * M1;    // [T,HFF] 8M floats
  float* fb = ws + 18 * M1;    // [T,HFF]

  embed_kernel<<<(T_ * C_) / 256, 256, 0, stream>>>(tokens, emb, x);

  for (int l = 0; l < L_; ++l) {
    const float* wq = Wq + (size_t)l * C_ * C_;
    const float* wk = Wk + (size_t)l * KVD_ * C_;
    const float* wv_ = Wv + (size_t)l * KVD_ * C_;
    const float* wo = Wo + (size_t)l * C_ * C_;
    const float* w1 = W1 + (size_t)l * HFF_ * C_;
    const float* w2 = W2 + (size_t)l * HFF_ * C_;
    const float* w3 = W3 + (size_t)l * C_ * HFF_;

    rmsnorm_kernel<<<T_, 256, 0, stream>>>(x, attn_norm_w + (size_t)l * C_, xn);

    gemm_nt_kernel<<<dim3(C_ / 128, T_ / 128), 256, 0, stream>>>(
        xn, wq, nullptr, q, T_, C_, C_);
    gemm_nt_kernel<<<dim3(KVD_ / 128, T_ / 128), 256, 0, stream>>>(
        xn, wk, nullptr, kb, T_, KVD_, C_);
    gemm_nt_kernel<<<dim3(KVD_ / 128, T_ / 128), 256, 0, stream>>>(
        xn, wv_, nullptr, vb, T_, KVD_, C_);

    rope_kernel<<<(T_ * H_ * 32) / 256, 256, 0, stream>>>(q, H_);
    rope_kernel<<<(T_ * HKV_ * 32) / 256, 256, 0, stream>>>(kb, HKV_);

    attn_kernel<<<dim3(T_ / 64, H_), 128, 0, stream>>>(q, kb, vb, y);

    // x = x + y @ Wo^T   (residual fused)
    gemm_nt_kernel<<<dim3(C_ / 128, T_ / 128), 256, 0, stream>>>(
        y, wo, x, x, T_, C_, C_);

    rmsnorm_kernel<<<T_, 256, 0, stream>>>(x, ff_norm_w + (size_t)l * C_, xn);

    gemm_nt_kernel<<<dim3(HFF_ / 128, T_ / 128), 256, 0, stream>>>(
        xn, w1, nullptr, fa, T_, HFF_, C_);
    gemm_nt_kernel<<<dim3(HFF_ / 128, T_ / 128), 256, 0, stream>>>(
        xn, w2, nullptr, fb, T_, HFF_, C_);

    swiglu_kernel<<<(T_ * HFF_) / 256, 256, 0, stream>>>(fa, fb);

    // x = x + h @ W3^T   (residual fused, K = HFF)
    gemm_nt_kernel<<<dim3(C_ / 128, T_ / 128), 256, 0, stream>>>(
        fa, w3, x, x, T_, C_, HFF_);
  }

  rmsnorm_kernel<<<T_, 256, 0, stream>>>(x, norm_f_w, xn);

  // logits = xn @ emb^T  -> [T, V]
  gemm_nt_kernel<<<dim3(V_ / 128, T_ / 128), 256, 0, stream>>>(
      xn, emb, nullptr, out, T_, V_, C_);
}
